// TransformerBlock_55087250539088
// MI455X (gfx1250) — matrix operand order repaired and checked by reading, not compiled, not run on hardware
//
#include <hip/hip_runtime.h>
#include <hip/hip_bf16.h>
#include <math.h>

// ---------------------------------------------------------------------------
// Transformer block for MI455X (gfx1250, wave32, WMMA).
// bf16 WMMA (16x16x32) for all GEMMs + flash attention, fp32 accumulate.
// Sparse top-2 MoE via token grouping (per-expert segments, capacity 2048).
// A-tiles staged via Tensor Data Mover (tensor_load_to_lds) when available.
// Workspace requirement: ~181 MB.
// ---------------------------------------------------------------------------

typedef __bf16 bf16;
typedef __attribute__((ext_vector_type(16))) __bf16 v16bf;
typedef __attribute__((ext_vector_type(8)))  float  v8f;

union Frag { v16bf v; uint4 u[2]; };

__device__ __forceinline__ v8f wmma_bf16(v16bf a, v16bf b, v8f c) {
  return __builtin_amdgcn_wmma_f32_16x16x32_bf16(false, a, false, b, (short)0, c,
                                                 false, false);
}

static const int cB = 2, cS = 1024, cD = 2048, cH = 16, cG = 4, cDK = 128,
                 cE = 8, cF = 4096, cT = 2048, cCAP = 2048;

#define GM 128
#define GN 128
#define GK 32
#define LSTR 48   // halves per LDS row (rows 96B -> 16B aligned frag loads)

// ---------------- Tensor Data Mover staging for A tiles ----------------
#if defined(__has_builtin)
#if __has_builtin(__builtin_amdgcn_tensor_load_to_lds)
#define HAVE_TDM 1
#endif
#endif

#ifdef HAVE_TDM
typedef __attribute__((ext_vector_type(4))) unsigned int tdm_v4u;
typedef __attribute__((ext_vector_type(8))) int tdm_v8i;
typedef __attribute__((ext_vector_type(4))) int tdm_v4i;

// 2D tile: GM rows x GK bf16 elements, global row stride = ld_elems.
// LDS destination padded: 16 DWORDs data then 8 DWORDs pad -> LSTR-half rows.
__device__ __forceinline__ void tdm_load_a_tile(const bf16* gsrc,
                                                unsigned lds_byte_addr,
                                                int ld_elems) {
  unsigned long long ga = (unsigned long long)(uintptr_t)gsrc;
  tdm_v4u g0;
  g0.x = 1u;                                     // count=1, user mode
  g0.y = lds_byte_addr;                          // lds_addr
  g0.z = (unsigned)(ga & 0xffffffffu);           // global_addr[31:0]
  g0.w = (unsigned)((ga >> 32) & 0x01ffffffu)    // global_addr[56:32]
         | (2u << 30);                           // type=2 ("image")
  tdm_v8i g1;
  // data_size=1 (2B) | pad_enable | pad_interval=3 (16 DW) | pad_amount=7 (8 DW)
  g1[0] = (1 << 16) | (1 << 20) | (3 << 22) | (7 << 25);
  g1[1] = (int)((unsigned)GK << 16);             // tensor_dim0[15:0] in [31:16]
  g1[2] = (int)((unsigned)GM << 16);             // dim0 hi=0 | tensor_dim1 lo16
  g1[3] = (int)((unsigned)GK << 16);             // dim1 hi=0 | tile_dim0=GK
  g1[4] = GM;                                    // tile_dim1=GM, tile_dim2=0
  g1[5] = ld_elems;                              // tensor_dim0_stride lo32
  g1[6] = 0;                                     // stride hi | dim1_stride lo
  g1[7] = 0;
  tdm_v4i z = {0, 0, 0, 0};
#if __clang_major__ >= 23
  tdm_v8i z8 = {0, 0, 0, 0, 0, 0, 0, 0};
  __builtin_amdgcn_tensor_load_to_lds(g0, g1, z, z, z8, 0);
#else
  __builtin_amdgcn_tensor_load_to_lds(g0, g1, z, z, 0);
#endif
#if __has_builtin(__builtin_amdgcn_s_wait_tensorcnt)
  __builtin_amdgcn_s_wait_tensorcnt(0);
#else
  asm volatile("s_wait_tensorcnt 0x0" ::: "memory");
#endif
}
#endif  // HAVE_TDM

// ---------------- RMSNorm (fp32 in -> bf16 out) ----------------
__global__ __launch_bounds__(256) void k_rmsnorm(const float* __restrict__ x,
                                                 const float* __restrict__ w,
                                                 bf16* __restrict__ o, int Dd) {
  int t = blockIdx.x;
  const float* xr = x + (size_t)t * Dd;
  float ss = 0.f;
  for (int d = threadIdx.x; d < Dd; d += 256) { float v = xr[d]; ss += v * v; }
#pragma unroll
  for (int off = 16; off; off >>= 1) ss += __shfl_xor(ss, off, 32);
  __shared__ float red[8];
  int wid = threadIdx.x >> 5;
  if ((threadIdx.x & 31) == 0) red[wid] = ss;
  __syncthreads();
  float tot = 0.f;
#pragma unroll
  for (int i = 0; i < 8; ++i) tot += red[i];
  float inv = rsqrtf(tot / (float)Dd + 1e-6f);
  bf16* orow = o + (size_t)t * Dd;
  for (int d = threadIdx.x; d < Dd; d += 256)
    orow[d] = (bf16)(xr[d] * inv * w[d]);
}

// ---------------- RoPE (in-place on bf16 [T][nh][128]) ----------------
__global__ void k_rope(bf16* __restrict__ qk, int nh) {
  int idx = blockIdx.x * blockDim.x + threadIdx.x;
  int total = cT * nh * 64;
  if (idx >= total) return;
  int i = idx & 63;
  int rem = idx >> 6;
  int hh = rem % nh;
  int t = rem / nh;
  int s = t & (cS - 1);
  float ang = (float)s * __powf(10000.f, -(float)i * (1.f / 64.f));
  float sn, cs;
  __sincosf(ang, &sn, &cs);
  size_t base = ((size_t)t * nh + hh) * cDK;
  float x1 = (float)qk[base + i], x2 = (float)qk[base + 64 + i];
  qk[base + i]      = (bf16)(x1 * cs - x2 * sn);
  qk[base + 64 + i] = (bf16)(x1 * sn + x2 * cs);
}

// ---------------- shared GEMM pieces ----------------
__device__ __forceinline__ void ld_A_direct(const bf16* __restrict__ A, int lda,
                                            int m0, int k0, bf16* As) {
  int t = threadIdx.x;
  int r = t >> 1, seg = (t & 1) * 16;
  const uint4* s = (const uint4*)(A + (size_t)(m0 + r) * lda + k0 + seg);
  uint4 d0 = s[0], d1 = s[1];
  bf16* dst = As + r * LSTR + seg;
  *(uint4*)dst = d0;
  *(uint4*)(dst + 8) = d1;
}

// Stage A tile: TDM path (wave 0 issues DMA) or manual fallback.
__device__ __forceinline__ void stage_A(const bf16* __restrict__ A, int lda,
                                        int m0, int k0, bf16* As, int wid) {
#ifdef HAVE_TDM
  if (wid == 0)
    tdm_load_a_tile(A + (size_t)m0 * lda + k0, (unsigned)(uintptr_t)(void*)As,
                    lda);
#else
  ld_A_direct(A, lda, m0, k0, As);
#endif
}

__device__ __forceinline__ void ld_B_cvt(const float* __restrict__ Bw, int ldb,
                                         int k0, int n0, bf16* Bs) {
  int t = threadIdx.x;
#pragma unroll
  for (int i = 0; i < 4; ++i) {
    int lin = t + i * 256;
    int kk = lin >> 5;
    int nn = (lin & 31) * 4;
    const float* src = Bw + (size_t)(k0 + kk) * ldb + n0 + nn;
    __builtin_prefetch(src + (size_t)GK * ldb, 0, 1);  // global_prefetch next tile
    const float4 f = *(const float4*)src;
    Bs[(nn + 0) * LSTR + kk] = (bf16)f.x;
    Bs[(nn + 1) * LSTR + kk] = (bf16)f.y;
    Bs[(nn + 2) * LSTR + kk] = (bf16)f.z;
    Bs[(nn + 3) * LSTR + kk] = (bf16)f.w;
  }
}

__device__ __forceinline__ void mma_step(const bf16* As, const bf16* Bs, int wm,
                                         int wn, int l16, int lh, v8f acc[2][4]) {
  Frag af[2], bfr[4];
#pragma unroll
  for (int mi = 0; mi < 2; ++mi) {
    const bf16* p = As + (wm + mi * 16 + l16) * LSTR + (lh ? 8 : 0);
    af[mi].u[0] = *(const uint4*)p;
    af[mi].u[1] = *(const uint4*)(p + 16);
  }
#pragma unroll
  for (int ni = 0; ni < 4; ++ni) {
    const bf16* p = Bs + (wn + ni * 16 + l16) * LSTR + (lh ? 8 : 0);
    bfr[ni].u[0] = *(const uint4*)p;
    bfr[ni].u[1] = *(const uint4*)(p + 16);
  }
#pragma unroll
  for (int mi = 0; mi < 2; ++mi)
#pragma unroll
    for (int ni = 0; ni < 4; ++ni)
      acc[mi][ni] = wmma_bf16(af[mi].v, bfr[ni].v, acc[mi][ni]);
}

// ---------------- GEMM: C(bf16) = A(bf16) * B(f32->bf16) ----------------
__global__ __launch_bounds__(256) void k_gemm_bf16out(
    const bf16* __restrict__ A, const float* __restrict__ Bw,
    bf16* __restrict__ C, int Kd, int N) {
  __shared__ bf16 As[GM * LSTR];
  __shared__ bf16 Bs[GN * LSTR];
  int m0 = blockIdx.x * GM, n0 = blockIdx.y * GN;
  int t = threadIdx.x, wid = t >> 5, lane = t & 31, l16 = lane & 15,
      lh = lane >> 4;
  int wm = (wid & 3) * 32, wn = (wid >> 2) * 64;
  v8f acc[2][4] = {};
  for (int k0 = 0; k0 < Kd; k0 += GK) {
    stage_A(A, Kd, m0, k0, As, wid);
    ld_B_cvt(Bw, N, k0, n0, Bs);
    __syncthreads();
    mma_step(As, Bs, wm, wn, l16, lh, acc);
    __syncthreads();
  }
#pragma unroll
  for (int mi = 0; mi < 2; ++mi)
#pragma unroll
    for (int ni = 0; ni < 4; ++ni)
#pragma unroll
      for (int r = 0; r < 8; ++r) {
        int row = m0 + wm + mi * 16 + r + lh * 8;
        int col = n0 + wn + ni * 16 + l16;
        C[(size_t)row * N + col] = (bf16)acc[mi][ni][r];
      }
}

// ---------------- GEMM + residual: X2 = resid + A*B ; Out = X2 ----------------
__global__ __launch_bounds__(256) void k_gemm_resid(
    const bf16* __restrict__ A, const float* __restrict__ Bw,
    const float* __restrict__ resid, float* __restrict__ X2,
    float* __restrict__ Out, int Kd, int N) {
  __shared__ bf16 As[GM * LSTR];
  __shared__ bf16 Bs[GN * LSTR];
  int m0 = blockIdx.x * GM, n0 = blockIdx.y * GN;
  int t = threadIdx.x, wid = t >> 5, lane = t & 31, l16 = lane & 15,
      lh = lane >> 4;
  int wm = (wid & 3) * 32, wn = (wid >> 2) * 64;
  v8f acc[2][4] = {};
  for (int k0 = 0; k0 < Kd; k0 += GK) {
    stage_A(A, Kd, m0, k0, As, wid);
    ld_B_cvt(Bw, N, k0, n0, Bs);
    __syncthreads();
    mma_step(As, Bs, wm, wn, l16, lh, acc);
    __syncthreads();
  }
#pragma unroll
  for (int mi = 0; mi < 2; ++mi)
#pragma unroll
    for (int ni = 0; ni < 4; ++ni)
#pragma unroll
      for (int r = 0; r < 8; ++r) {
        int row = m0 + wm + mi * 16 + r + lh * 8;
        int col = n0 + wn + ni * 16 + l16;
        size_t idx = (size_t)row * N + col;
        float v = resid[idx] + acc[mi][ni][r];
        X2[idx] = v;
        Out[idx] = v;
      }
}

// ---------------- Flash attention (causal, GQA) ----------------
#define AKS 144  // halves per K-chunk LDS row (288B)
#define AVS 48   // halves per V^T LDS row (96B)

__global__ __launch_bounds__(256) void k_attn(const bf16* __restrict__ Q,
                                              const bf16* __restrict__ Kb,
                                              const bf16* __restrict__ Vb,
                                              bf16* __restrict__ O) {
  __shared__ bf16 Kc[32 * AKS];
  __shared__ bf16 Vt[128 * AVS];
  __shared__ bf16 Pb[8][16 * 32];
  int q0 = blockIdx.x * 128;
  int h = blockIdx.y, bb = blockIdx.z;
  int g = h >> 2;  // H/G = 4
  int t = threadIdx.x, wid = t >> 5, lane = t & 31, l16 = lane & 15,
      lh = lane >> 4;
  int qbase = q0 + wid * 16;

  Frag qa[4];
  {
    const bf16* qp = Q + (((size_t)(bb * cS + qbase + l16)) * cH + h) * cDK;
#pragma unroll
    for (int kd = 0; kd < 4; ++kd) {
      const bf16* p = qp + kd * 32 + (lh ? 8 : 0);
      qa[kd].u[0] = *(const uint4*)p;
      qa[kd].u[1] = *(const uint4*)(p + 16);
    }
  }
  v8f oacc[8] = {};
  float mrow[8], lrow[8];
#pragma unroll
  for (int r = 0; r < 8; ++r) { mrow[r] = -1e30f; lrow[r] = 0.f; }

  int nchunk = (q0 + 128) >> 5;
  for (int c = 0; c < nchunk; ++c) {
    int k0 = c * 32;
    {  // cooperative stage: K chunk row-major, V chunk transposed
      int row = t >> 3, seg = (t & 7) * 16;
      const bf16* ks = Kb + (((size_t)(bb * cS + k0 + row)) * cG + g) * cDK + seg;
      uint4 d0 = ((const uint4*)ks)[0], d1 = ((const uint4*)ks)[1];
      bf16* dst = Kc + row * AKS + seg;
      *(uint4*)dst = d0;
      *(uint4*)(dst + 8) = d1;
      const bf16* vs = Vb + (((size_t)(bb * cS + k0 + row)) * cG + g) * cDK + seg;
#pragma unroll
      for (int j = 0; j < 16; ++j) Vt[(seg + j) * AVS + row] = vs[j];
    }
    __syncthreads();
    if (k0 <= qbase + 15) {  // uniform per wave
      v8f sacc[2] = {};
#pragma unroll
      for (int tt = 0; tt < 2; ++tt)
#pragma unroll
        for (int kd = 0; kd < 4; ++kd) {
          Frag kf;
          const bf16* p = Kc + (tt * 16 + l16) * AKS + kd * 32 + (lh ? 8 : 0);
          kf.u[0] = *(const uint4*)p;
          kf.u[1] = *(const uint4*)(p + 16);
          sacc[tt] = wmma_bf16(qa[kd].v, kf.v, sacc[tt]);
        }
      float pv[2][8], mch[8];
#pragma unroll
      for (int r = 0; r < 8; ++r) mch[r] = -1e30f;
#pragma unroll
      for (int tt = 0; tt < 2; ++tt)
#pragma unroll
        for (int r = 0; r < 8; ++r) {
          float v = sacc[tt][r] * 0.08838834764831845f;  // 1/sqrt(128)
          int kg = k0 + tt * 16 + l16;
          int qg = qbase + r + lh * 8;
          v = (kg > qg) ? -1e30f : v;
          pv[tt][r] = v;
          mch[r] = fmaxf(mch[r], v);
        }
#pragma unroll
      for (int r = 0; r < 8; ++r) {
#pragma unroll
        for (int off = 1; off < 16; off <<= 1)
          mch[r] = fmaxf(mch[r], __shfl_xor(mch[r], off, 32));
        float mn = fmaxf(mrow[r], mch[r]);
        float al = __expf(mrow[r] - mn);
        mrow[r] = mn;
        lrow[r] *= al;
#pragma unroll
        for (int dn = 0; dn < 8; ++dn) oacc[dn][r] *= al;
      }
      float rs[8] = {};
#pragma unroll
      for (int tt = 0; tt < 2; ++tt)
#pragma unroll
        for (int r = 0; r < 8; ++r) {
          float p = __expf(pv[tt][r] - mrow[r]);
          pv[tt][r] = p;
          rs[r] += p;
        }
#pragma unroll
      for (int r = 0; r < 8; ++r) {
#pragma unroll
        for (int off = 1; off < 16; off <<= 1) rs[r] += __shfl_xor(rs[r], off, 32);
        lrow[r] += rs[r];
      }
      // bounce P through LDS: C-layout -> A-layout
      bf16* pb = &Pb[wid][0];
#pragma unroll
      for (int tt = 0; tt < 2; ++tt)
#pragma unroll
        for (int r = 0; r < 8; ++r)
          pb[(r + lh * 8) * 32 + tt * 16 + l16] = (bf16)pv[tt][r];
      asm volatile("s_wait_dscnt 0" ::: "memory");  // intra-wave LDS fence
      Frag pa;
      {
        const bf16* p = pb + l16 * 32 + (lh ? 8 : 0);
        pa.u[0] = *(const uint4*)p;
        pa.u[1] = *(const uint4*)(p + 16);
      }
#pragma unroll
      for (int dn = 0; dn < 8; ++dn) {
        Frag vf;
        const bf16* p = Vt + (dn * 16 + l16) * AVS + (lh ? 8 : 0);
        vf.u[0] = *(const uint4*)p;
        vf.u[1] = *(const uint4*)(p + 16);
        oacc[dn] = wmma_bf16(pa.v, vf.v, oacc[dn]);
      }
    }
    __syncthreads();
  }
#pragma unroll
  for (int dn = 0; dn < 8; ++dn)
#pragma unroll
    for (int r = 0; r < 8; ++r) {
      int row = qbase + r + lh * 8;
      float v = oacc[dn][r] / lrow[r];
      O[(((size_t)(bb * cS + row)) * cH + h) * cDK + dn * 16 + l16] = (bf16)v;
    }
}

// ---------------- Router: logits, top-2, gates, expert grouping ----------------
__global__ void k_zero_cursor(int* c) {
  if (threadIdx.x < cE) c[threadIdx.x] = 0;
}

__global__ __launch_bounds__(256) void k_router(const bf16* __restrict__ H2,
                                                const float* __restrict__ Wr,
                                                int* __restrict__ cursor,
                                                int* __restrict__ perm,
                                                float* __restrict__ pgate) {
  int tk = blockIdx.x;
  const bf16* hr = H2 + (size_t)tk * cD;
  float acc[8] = {};
  for (int d = threadIdx.x; d < cD; d += 256) {
    float hv = (float)hr[d];
    const float* wr = Wr + (size_t)d * cE;
#pragma unroll
    for (int e = 0; e < 8; ++e) acc[e] += hv * wr[e];
  }
#pragma unroll
  for (int e = 0; e < 8; ++e)
#pragma unroll
    for (int off = 16; off; off >>= 1) acc[e] += __shfl_xor(acc[e], off, 32);
  __shared__ float red[8][8];
  int wid = threadIdx.x >> 5;
  if ((threadIdx.x & 31) == 0)
    for (int e = 0; e < 8; ++e) red[wid][e] = acc[e];
  __syncthreads();
  if (threadIdx.x == 0) {
    float lg[8];
    for (int e = 0; e < 8; ++e) {
      float s = 0.f;
      for (int w = 0; w < 8; ++w) s += red[w][e];
      lg[e] = s;
    }
    int i0 = 0;
    for (int e = 1; e < 8; ++e)
      if (lg[e] > lg[i0]) i0 = e;
    int i1 = (i0 == 0) ? 1 : 0;
    for (int e = 0; e < 8; ++e)
      if (e != i0 && lg[e] > lg[i1]) i1 = e;
    float ex = __expf(lg[i1] - lg[i0]);
    float g0 = 1.f / (1.f + ex), g1 = ex / (1.f + ex);
    int p0 = atomicAdd(cursor + i0, 1);
    perm[i0 * cCAP + p0] = tk;
    pgate[i0 * cCAP + p0] = g0;
    int p1 = atomicAdd(cursor + i1, 1);
    perm[i1 * cCAP + p1] = tk;
    pgate[i1 * cCAP + p1] = g1;
  }
}

// ---------------- MoE GEMM1: mid = silu(h2[tok] @ W1[e]) ----------------
__global__ __launch_bounds__(256) void k_moe_gemm1(
    const bf16* __restrict__ H2, const float* __restrict__ W1,
    const int* __restrict__ perm, const int* __restrict__ cnt,
    bf16* __restrict__ Mid) {
  int e = (int)((blockIdx.x * GM) / cCAP);
  int ls0 = blockIdx.x * GM - e * cCAP;
  int count = cnt[e];
  if (ls0 >= count) return;
  const float* Bw = W1 + (size_t)e * cD * cF;
  int n0 = blockIdx.y * GN;
  __shared__ bf16 As[GM * LSTR];
  __shared__ bf16 Bs[GN * LSTR];
  __shared__ int toks[GM];
  int t = threadIdx.x, wid = t >> 5, lane = t & 31, l16 = lane & 15,
      lh = lane >> 4;
  int wm = (wid & 3) * 32, wn = (wid >> 2) * 64;
  if (t < GM) {
    int ls = ls0 + t;
    toks[t] = (ls < count) ? perm[e * cCAP + ls] : -1;
  }
  __syncthreads();
  v8f acc[2][4] = {};
  for (int k0 = 0; k0 < cD; k0 += GK) {
    {  // gathered A load
      int r = t >> 1, seg = (t & 1) * 16;
      int tok = toks[r];
      uint4 d0 = {0, 0, 0, 0}, d1 = {0, 0, 0, 0};
      if (tok >= 0) {
        const uint4* s = (const uint4*)(H2 + (size_t)tok * cD + k0 + seg);
        d0 = s[0];
        d1 = s[1];
      }
      bf16* dst = As + r * LSTR + seg;
      *(uint4*)dst = d0;
      *(uint4*)(dst + 8) = d1;
    }
    ld_B_cvt(Bw, cF, k0, n0, Bs);
    __syncthreads();
    mma_step(As, Bs, wm, wn, l16, lh, acc);
    __syncthreads();
  }
#pragma unroll
  for (int mi = 0; mi < 2; ++mi)
#pragma unroll
    for (int ni = 0; ni < 4; ++ni)
#pragma unroll
      for (int r = 0; r < 8; ++r) {
        int lr = ls0 + wm + mi * 16 + r + lh * 8;
        if (lr < count) {
          float v = acc[mi][ni][r];
          v = v / (1.f + __expf(-v));  // silu
          Mid[(size_t)(e * cCAP + lr) * cF + n0 + wn + ni * 16 + l16] = (bf16)v;
        }
      }
}

// ---------------- MoE GEMM2: Out[tok] += gate * (mid @ W2[e]) ----------------
__global__ __launch_bounds__(256) void k_moe_gemm2(
    const bf16* __restrict__ Mid, const float* __restrict__ W2,
    const int* __restrict__ perm, const float* __restrict__ pgate,
    const int* __restrict__ cnt, float* __restrict__ Out) {
  int e = (int)((blockIdx.x * GM) / cCAP);
  int ls0 = blockIdx.x * GM - e * cCAP;
  int count = cnt[e];
  if (ls0 >= count) return;
  const float* Bw = W2 + (size_t)e * cF * cD;
  int m0 = blockIdx.x * GM;  // slot base
  int n0 = blockIdx.y * GN;
  __shared__ bf16 As[GM * LSTR];
  __shared__ bf16 Bs[GN * LSTR];
  int t = threadIdx.x, wid = t >> 5, lane = t & 31, l16 = lane & 15,
      lh = lane >> 4;
  int wm = (wid & 3) * 32, wn = (wid >> 2) * 64;
  v8f acc[2][4] = {};
  for (int k0 = 0; k0 < cF; k0 += GK) {
    stage_A(Mid, cF, m0, k0, As, wid);
    ld_B_cvt(Bw, cD, k0, n0, Bs);
    __syncthreads();
    mma_step(As, Bs, wm, wn, l16, lh, acc);
    __syncthreads();
  }
#pragma unroll
  for (int mi = 0; mi < 2; ++mi)
#pragma unroll
    for (int ni = 0; ni < 4; ++ni)
#pragma unroll
      for (int r = 0; r < 8; ++r) {
        int lr = ls0 + wm + mi * 16 + r + lh * 8;
        if (lr < count) {
          int slot = e * cCAP + lr;
          int tok = perm[slot];
          float g = pgate[slot];
          atomicAdd(Out + (size_t)tok * cD + n0 + wn + ni * 16 + l16,
                    g * acc[mi][ni][r]);
        }
      }
}

// ---------------------------------------------------------------------------
extern "C" void kernel_launch(void* const* d_in, const int* in_sizes, int n_in,
                              void* d_out, int out_size, void* d_ws,
                              size_t ws_size, hipStream_t stream) {
  const float* x    = (const float*)d_in[0];
  const float* ln1w = (const float*)d_in[1];
  const float* Wq   = (const float*)d_in[2];
  const float* Wk   = (const float*)d_in[3];
  const float* Wv   = (const float*)d_in[4];
  const float* Wo   = (const float*)d_in[5];
  const float* ln2w = (const float*)d_in[6];
  const float* Wr   = (const float*)d_in[7];
  const float* W1   = (const float*)d_in[8];
  const float* W2   = (const float*)d_in[9];
  float* Out = (float*)d_out;
  (void)in_sizes; (void)n_in; (void)out_size; (void)ws_size;

  char* base = (char*)d_ws;
  size_t off = 0;
  auto alloc = [&](size_t bytes) -> void* {
    void* p = base + off;
    off = (off + bytes + 255) & ~(size_t)255;
    return p;
  };
  bf16*  h1   = (bf16*)alloc((size_t)cT * cD * 2);
  bf16*  qb   = (bf16*)alloc((size_t)cT * cH * cDK * 2);
  bf16*  kb   = (bf16*)alloc((size_t)cT * cG * cDK * 2);
  bf16*  vb   = (bf16*)alloc((size_t)cT * cG * cDK * 2);
  bf16*  attn = (bf16*)alloc((size_t)cT * cH * cDK * 2);
  float* x2   = (float*)alloc((size_t)cT * cD * 4);
  bf16*  h2   = (bf16*)alloc((size_t)cT * cD * 2);
  int*   perm = (int*)alloc((size_t)cE * cCAP * 4);
  float* pg   = (float*)alloc((size_t)cE * cCAP * 4);
  int*   cur  = (int*)alloc(256);
  bf16*  mid  = (bf16*)alloc((size_t)cE * cCAP * cF * 2);

  // 1) h1 = rmsnorm(x) in bf16
  k_rmsnorm<<<cT, 256, 0, stream>>>(x, ln1w, h1, cD);
  // 2) q/k/v projections
  k_gemm_bf16out<<<dim3(cT / GM, (cH * cDK) / GN), 256, 0, stream>>>(
      h1, Wq, qb, cD, cH * cDK);
  k_gemm_bf16out<<<dim3(cT / GM, (cG * cDK) / GN), 256, 0, stream>>>(
      h1, Wk, kb, cD, cG * cDK);
  k_gemm_bf16out<<<dim3(cT / GM, (cG * cDK) / GN), 256, 0, stream>>>(
      h1, Wv, vb, cD, cG * cDK);
  // 3) RoPE
  k_rope<<<(cT * cH * 64) / 256, 256, 0, stream>>>(qb, cH);
  k_rope<<<(cT * cG * 64) / 256, 256, 0, stream>>>(kb, cG);
  // 4) causal flash attention (GQA)
  k_attn<<<dim3(cS / 128, cH, cB), 256, 0, stream>>>(qb, kb, vb, attn);
  // 5) out-proj + residual -> x2 (ws) and Out (pre-fill for MoE scatter)
  k_gemm_resid<<<dim3(cT / GM, cD / GN), 256, 0, stream>>>(attn, Wo, x, x2, Out,
                                                           cH * cDK, cD);
  // 6) h2 = rmsnorm(x2)
  k_rmsnorm<<<cT, 256, 0, stream>>>(x2, ln2w, h2, cD);
  // 7) routing + expert grouping
  k_zero_cursor<<<1, 32, 0, stream>>>(cur);
  k_router<<<cT, 256, 0, stream>>>(h2, Wr, cur, perm, pg);
  // 8) grouped MoE GEMMs (sparse: only routed tokens per expert)
  k_moe_gemm1<<<dim3((cE * cCAP) / GM, cF / GN), 256, 0, stream>>>(h2, W1, perm,
                                                                   cur, mid);
  k_moe_gemm2<<<dim3((cE * cCAP) / GM, cD / GN), 256, 0, stream>>>(
      mid, W2, perm, pg, cur, Out);
}
